// NestedPPGN_7842610283378
// MI455X (gfx1250) — matrix operand order repaired and checked by reading, not compiled, not run on hardware
//
#include <hip/hip_runtime.h>
#include <stdint.h>

// ---------------------------------------------------------------------------
// Types for CDNA5 WMMA (wave32, v_wmma_f32_16x16x32_bf16)
// ---------------------------------------------------------------------------
typedef __bf16 bf16;
typedef __attribute__((ext_vector_type(16))) __bf16 v16bf;
typedef __attribute__((ext_vector_type(8)))  float  v8f;

// ---------------------------------------------------------------------------
// Problem dimensions
// ---------------------------------------------------------------------------
#define GG      16
#define SS      30
#define KK      25
#define BSUB    480              // G*S
#define NNODES  12000            // BSUB*KK
#define ELOC    48000
#define EGLB    1920
#define NCHUNK  4
#define NBC     (BSUB / NCHUNK)  // 120 subgraphs per chunk
#define PC      (NBC * KK * KK)  // 75000 pixels per chunk
#define MPADL   75008            // PC rounded up to 32
#define PG      (GG * SS * SS)   // 14400 (multiple of 32)

static inline int divup(int a, int b) { return (a + b - 1) / b; }

// ---------------------------------------------------------------------------
// Fragment loaders.
// A (16x32 bf16): lane L holds row M=L%16; element e<8 -> K = 8*(L/16)+e,
//                 e>=8 -> K = 16 + 8*(L/16) + (e-8).  => two contiguous 16B runs.
// B (32x16 bf16): lane L holds column N=L%16 (= row of W, K-contiguous);
//                 same K mapping as A.                 => two contiguous 16B runs.
// ---------------------------------------------------------------------------
__device__ __forceinline__ v16bf load_a_frag(const bf16* row, int k0, int half) {
  union { v16bf v; uint4 q[2]; } u;
  u.q[0] = *(const uint4*)(row + k0 + 8 * half);
  u.q[1] = *(const uint4*)(row + k0 + 16 + 8 * half);
  return u.v;
}
__device__ __forceinline__ v16bf load_b_frag(const bf16* row, int k0, int half) {
  union { v16bf v; uint4 q[2]; } u;
  u.q[0] = *(const uint4*)(row + k0 + 8 * half);
  u.q[1] = *(const uint4*)(row + k0 + 16 + 8 * half);
  return u.v;
}
__device__ __forceinline__ v8f wmma_bf16(v16bf a, v16bf b, v8f c) {
  return __builtin_amdgcn_wmma_f32_16x16x32_bf16(false, a, false, b,
                                                 (short)0, c, false, false);
}

// Epilogue store shared by both GEMM kernels.
// storeMode: 0 bf16 row-major [M,N]; 1 f32 row-major [M,N];
//            2 bf16 bmm-A layout [(b*Co+n)][32][32] at (i,j)   (m1: k=i, n=j)
//            3 bf16 bmm-B layout [(b*Co+n)][32][32] at (j,i)   (m2^T: rows=m)
//            4 f32 column-0 only -> out[m]  (final head)
__device__ __forceinline__ void store_elem(void* out, float val, int m, int col,
                                           int N, int storeMode, int nn, int Co) {
  switch (storeMode) {
    case 0: ((bf16*)out)[(size_t)m * N + col] = (bf16)val; break;
    case 1: ((float*)out)[(size_t)m * N + col] = val; break;
    case 2: { int sp = m % (nn * nn); int bl = m / (nn * nn);
              int i = sp / nn, j = sp % nn;
              ((bf16*)out)[(((size_t)bl * Co + col) * 32 + i) * 32 + j] = (bf16)val; } break;
    case 3: { int sp = m % (nn * nn); int bl = m / (nn * nn);
              int i = sp / nn, j = sp % nn;
              ((bf16*)out)[(((size_t)bl * Co + col) * 32 + j) * 32 + i] = (bf16)val; } break;
    case 4: if (col == 0) ((float*)out)[m] = val; break;
  }
}

// ---------------------------------------------------------------------------
// Generic fused GEMM (one wave = one 16x16 tile), used for small head GEMMs:
//   out = act( A1[M,K1] * B1[N,K1]^T + A2[M,K2] * B2[N,K2]^T + bias )
// ---------------------------------------------------------------------------
__global__ __launch_bounds__(256) void k_gemm(
    const bf16* __restrict__ A1, int K1, const bf16* __restrict__ B1,
    const bf16* __restrict__ A2, const bf16* __restrict__ B2, int K2,
    const float* __restrict__ bias, void* __restrict__ out,
    int M, int N, int validM, int storeMode, int relu, int nn, int Co)
{
  int wid  = (blockIdx.x * blockDim.x + threadIdx.x) >> 5;
  int lane = threadIdx.x & 31;
  int ntn  = N >> 4;
  int T    = (M >> 4) * ntn;
  if (wid >= T) return;                 // wave-uniform: EXEC stays all-ones
  int tm = wid / ntn, tn = wid % ntn;
  int half = lane >> 4, l16 = lane & 15;
  int row = tm * 16 + l16;
  int col = tn * 16 + l16;

  v8f acc;
  float bv = bias ? bias[col] : 0.0f;
  for (int e = 0; e < 8; ++e) acc[e] = bv;

  const bf16* arow = A1 + (size_t)row * K1;
  const bf16* brow = B1 + (size_t)col * K1;
  for (int k0 = 0; k0 < K1; k0 += 32)
    acc = wmma_bf16(load_a_frag(arow, k0, half), load_b_frag(brow, k0, half), acc);
  if (K2 > 0) {
    const bf16* arow2 = A2 + (size_t)row * K2;
    const bf16* brow2 = B2 + (size_t)col * K2;
    for (int k0 = 0; k0 < K2; k0 += 32)
      acc = wmma_bf16(load_a_frag(arow2, k0, half), load_b_frag(brow2, k0, half), acc);
  }
  for (int v = 0; v < 8; ++v) {
    float val = acc[v];
    if (relu) val = fmaxf(val, 0.0f);
    int m = tm * 16 + v + 8 * half;
    if (m < validM) store_elem(out, val, m, col, N, storeMode, nn, Co);
  }
}

// ---------------------------------------------------------------------------
// 2x2 register-blocked GEMM: one wave = 32x32 output tile (M%32==0, N%32==0).
// 4 WMMAs per 4 fragment loads -> 2x arithmetic intensity vs 16x16 tiling,
// 4 independent accumulation chains for XDL pipelining.
// ---------------------------------------------------------------------------
__global__ __launch_bounds__(256) void k_gemm2(
    const bf16* __restrict__ A1, int K1, const bf16* __restrict__ B1,
    const bf16* __restrict__ A2, const bf16* __restrict__ B2, int K2,
    const float* __restrict__ bias, void* __restrict__ out,
    int M, int N, int validM, int storeMode, int relu, int nn, int Co)
{
  int wid  = (blockIdx.x * blockDim.x + threadIdx.x) >> 5;
  int lane = threadIdx.x & 31;
  int ntn  = N >> 5;
  int T    = (M >> 5) * ntn;
  if (wid >= T) return;
  int tm = wid / ntn, tn = wid % ntn;
  int half = lane >> 4, l16 = lane & 15;
  int row0 = tm * 32 + l16, row1 = row0 + 16;
  int col0 = tn * 32 + l16, col1 = col0 + 16;

  v8f acc[2][2];
  float bv0 = bias ? bias[col0] : 0.0f;
  float bv1 = bias ? bias[col1] : 0.0f;
  for (int e = 0; e < 8; ++e) {
    acc[0][0][e] = bv0; acc[1][0][e] = bv0;
    acc[0][1][e] = bv1; acc[1][1][e] = bv1;
  }
  const bf16* ar0 = A1 + (size_t)row0 * K1;
  const bf16* ar1 = A1 + (size_t)row1 * K1;
  const bf16* br0 = B1 + (size_t)col0 * K1;
  const bf16* br1 = B1 + (size_t)col1 * K1;
  for (int k0 = 0; k0 < K1; k0 += 32) {
    v16bf a0 = load_a_frag(ar0, k0, half);
    v16bf a1 = load_a_frag(ar1, k0, half);
    v16bf b0 = load_b_frag(br0, k0, half);
    v16bf b1 = load_b_frag(br1, k0, half);
    acc[0][0] = wmma_bf16(a0, b0, acc[0][0]);
    acc[0][1] = wmma_bf16(a0, b1, acc[0][1]);
    acc[1][0] = wmma_bf16(a1, b0, acc[1][0]);
    acc[1][1] = wmma_bf16(a1, b1, acc[1][1]);
  }
  if (K2 > 0) {
    const bf16* cr0 = A2 + (size_t)row0 * K2;
    const bf16* cr1 = A2 + (size_t)row1 * K2;
    const bf16* dr0 = B2 + (size_t)col0 * K2;
    const bf16* dr1 = B2 + (size_t)col1 * K2;
    for (int k0 = 0; k0 < K2; k0 += 32) {
      v16bf a0 = load_a_frag(cr0, k0, half);
      v16bf a1 = load_a_frag(cr1, k0, half);
      v16bf b0 = load_b_frag(dr0, k0, half);
      v16bf b1 = load_b_frag(dr1, k0, half);
      acc[0][0] = wmma_bf16(a0, b0, acc[0][0]);
      acc[0][1] = wmma_bf16(a0, b1, acc[0][1]);
      acc[1][0] = wmma_bf16(a1, b0, acc[1][0]);
      acc[1][1] = wmma_bf16(a1, b1, acc[1][1]);
    }
  }
  for (int sm = 0; sm < 2; ++sm)
    for (int sn = 0; sn < 2; ++sn) {
      int colx = tn * 32 + sn * 16 + l16;
      for (int v = 0; v < 8; ++v) {
        float val = acc[sm][sn][v];
        if (relu) val = fmaxf(val, 0.0f);
        int m = tm * 32 + sm * 16 + v + 8 * half;
        if (m < validM) store_elem(out, val, m, colx, N, storeMode, nn, Co);
      }
    }
}

// ---------------------------------------------------------------------------
// Batched per-(b,channel) NxN matmul:  mult[b,o,k,m] = sum_n m1[b,o,k,n]*m2[b,o,n,m]
// m1 stored [slot][k][n<=32]; m2 stored transposed [slot][m][n<=32]; slot=b*128+o.
// Output written pixel-major [P,128] so it feeds the skip GEMM directly.
// ---------------------------------------------------------------------------
__global__ __launch_bounds__(256) void k_bmm(
    const bf16* __restrict__ M1t, const bf16* __restrict__ M2t,
    bf16* __restrict__ Out, int nb, int nn)
{
  int wid  = (blockIdx.x * blockDim.x + threadIdx.x) >> 5;
  int lane = threadIdx.x & 31;
  int T = nb * 128 * 4;
  if (wid >= T) return;
  int t = wid & 3, bo = wid >> 2;
  int tm = t >> 1, tn = t & 1;
  int half = lane >> 4, l16 = lane & 15;
  const bf16* Abase = M1t + (size_t)bo * 1024;
  const bf16* Bbase = M2t + (size_t)bo * 1024;
  v16bf a = load_a_frag(Abase + (size_t)(tm * 16 + l16) * 32, 0, half);
  v16bf b = load_b_frag(Bbase + (size_t)(tn * 16 + l16) * 32, 0, half);
  v8f acc; for (int e = 0; e < 8; ++e) acc[e] = 0.0f;
  acc = wmma_bf16(a, b, acc);
  int bb = bo >> 7, o = bo & 127;
  int m = tn * 16 + l16;
  for (int v = 0; v < 8; ++v) {
    int k = tm * 16 + v + 8 * half;
    if (k < nn && m < nn) {
      size_t p = ((size_t)bb * nn + k) * nn + m;
      Out[p * 128 + o] = (bf16)acc[v];
    }
  }
}

// ---------------------------------------------------------------------------
// diag / off-diag max pool: one wave per (b, c). pooled[b, c]=diag max,
// pooled[b,128+c]=offdiag max (equivalent to reference's val*eye trick).
// ---------------------------------------------------------------------------
__global__ __launch_bounds__(256) void k_pool(
    const bf16* __restrict__ Z, bf16* __restrict__ pooled,
    int nb, int nn, int b0)
{
  int wid  = (blockIdx.x * blockDim.x + threadIdx.x) >> 5;
  int lane = threadIdx.x & 31;
  if (wid >= nb * 128) return;
  int bl = wid >> 7, c = wid & 127;
  float md = -3.4e38f, mo = -3.4e38f;
  int NN = nn * nn;
  for (int s = lane; s < NN; s += 32) {
    int i = s / nn, j = s % nn;
    float v = (float)Z[(((size_t)bl * nn + i) * nn + j) * 128 + c];
    if (i == j) md = fmaxf(md, v); else mo = fmaxf(mo, v);
  }
  for (int d = 16; d; d >>= 1) {
    md = fmaxf(md, __shfl_xor(md, d, 32));
    mo = fmaxf(mo, __shfl_xor(mo, d, 32));
  }
  if (lane == 0) {
    pooled[(size_t)(b0 + bl) * 256 + c]       = (bf16)md;
    pooled[(size_t)(b0 + bl) * 256 + 128 + c] = (bf16)mo;
  }
}

// ---------------------------------------------------------------------------
// Assemble zz input [P,32] bf16: ch0..4 dense edges (pre-transposed scatter),
// ch5 pairwise distance, ch6..6+F-1 diag node features, rest zero.
// ---------------------------------------------------------------------------
__global__ __launch_bounds__(256) void k_assemble(
    bf16* __restrict__ Z, const float* __restrict__ E5,
    const float* __restrict__ nodefeat, int F,
    const float* __restrict__ pos, int nn, int b0, int nbc)
{
  int p = blockIdx.x * blockDim.x + threadIdx.x;
  int Pcc = nbc * nn * nn;
  if (p >= Pcc) return;
  int NN = nn * nn;
  int bl = p / NN, sp = p % NN;
  int i = sp / nn, j = sp % nn;
  int b = b0 + bl;
  int ni = b * nn + i, nj = b * nn + j;
  float dx = pos[ni * 3 + 0] - pos[nj * 3 + 0];
  float dy = pos[ni * 3 + 1] - pos[nj * 3 + 1];
  float dz = pos[ni * 3 + 2] - pos[nj * 3 + 2];
  float d2 = dx * dx + dy * dy + dz * dz;
  float dist = d2 > 0.0f ? sqrtf(d2) : 0.0f;
  bf16* zr = Z + (size_t)p * 32;
  for (int ch = 0; ch < 5; ++ch) zr[ch] = (bf16)E5[(size_t)p * 5 + ch];
  zr[5] = (bf16)dist;
  for (int f = 0; f < F; ++f)
    zr[6 + f] = (i == j) ? (bf16)nodefeat[(size_t)ni * F + f] : (bf16)0.0f;
  for (int ch = 6 + F; ch < 32; ++ch) zr[ch] = (bf16)0.0f;
}

// Scatter-add edge attrs into transposed dense pixel grid (zz = transpose(0,3,2,1)).
__global__ __launch_bounds__(256) void k_scatter(
    float* __restrict__ E5, const int* __restrict__ eidx,
    const float* __restrict__ eattr, int E, int nn, int b0, int nbc)
{
  int t = blockIdx.x * blockDim.x + threadIdx.x;
  if (t >= E) return;
  int rown = eidx[t], coln = eidx[E + t];
  int b = rown / nn;
  if (b < b0 || b >= b0 + nbc) return;
  int r = rown % nn, c = coln % nn;
  size_t p = ((size_t)(b - b0) * nn + c) * nn + r;   // (i=c, j=r): transpose
  for (int k = 0; k < 4; ++k) atomicAdd(&E5[p * 5 + k], eattr[(size_t)t * 5 + k]);
  atomicAdd(&E5[p * 5 + 4], 1.0f);
}

// node features for local stage: [nt_emb + z_emb (8) | x (11)] -> f32 [N,19]
__global__ __launch_bounds__(256) void k_nodefeat(
    float* __restrict__ nf, const float* __restrict__ x,
    const float* __restrict__ ntemb, const float* __restrict__ zemb,
    const int* __restrict__ nt, const int* __restrict__ z, int n)
{
  int t = blockIdx.x * blockDim.x + threadIdx.x;
  if (t >= n) return;
  int ti = nt[t], zi = z[t];
  for (int f = 0; f < 8; ++f)
    nf[(size_t)t * 19 + f] = ntemb[ti * 8 + f] + zemb[zi * 8 + f];
  for (int f = 0; f < 11; ++f)
    nf[(size_t)t * 19 + 8 + f] = x[(size_t)t * 11 + f];
}

// f32 weight [Co,srcLd] (subrange colOff..colOff+K0) -> bf16 [Copad,Kpad], zero pad
__global__ __launch_bounds__(256) void k_wconv(
    bf16* __restrict__ dst, const float* __restrict__ src,
    int Co, int Copad, int K0, int Kpad, int colOff, int srcLd)
{
  int t = blockIdx.x * blockDim.x + threadIdx.x;
  if (t >= Copad * Kpad) return;
  int co = t / Kpad, k = t % Kpad;
  float v = (co < Co && k < K0) ? src[(size_t)co * srcLd + colOff + k] : 0.0f;
  dst[t] = (bf16)v;
}

__global__ __launch_bounds__(256) void k_zero32(unsigned* __restrict__ p, int n)
{
  int t = blockIdx.x * blockDim.x + threadIdx.x;
  if (t < n) p[t] = 0u;
}

__global__ __launch_bounds__(256) void k_padf32(
    float* __restrict__ dst, const float* __restrict__ src, int n0, int npad)
{
  int t = blockIdx.x * blockDim.x + threadIdx.x;
  if (t < npad) dst[t] = t < n0 ? src[t] : 0.0f;
}

// ---------------------------------------------------------------------------
// Host orchestration.
// Input layout (top-level dict order; nested `params` pytree-flattened with
// sorted dict keys, lists/tuples in order):
//   0 x, 1 pos, 2 edge_attr, 3 original_edge_attr, 4 original_pos,
//   5..54 params leaves:
//     0..9   global_blocks[0]: mlp1(W,b)x2, mlp2(W,b)x2, skip(W,b)
//     10..19 global_blocks[1]
//     20,21 global_fc1  22,23 global_fc2
//     24..33 local_blocks[0]  34..43 local_blocks[1]
//     44,45 local_fc1  46,47 local_fc2  48 nt_emb  49 z_emb
//   55 z, 56 node_type, 57 edge_index, 58 node_to_subgraph,
//   59 original_edge_index, 60 subgraph_to_graph
// ---------------------------------------------------------------------------
extern "C" void kernel_launch(void* const* d_in, const int* in_sizes, int n_in,
                              void* d_out, int out_size, void* d_ws, size_t ws_size,
                              hipStream_t stream)
{
  (void)in_sizes; (void)n_in; (void)out_size; (void)ws_size;
  const float* x      = (const float*)d_in[0];
  const float* pos    = (const float*)d_in[1];
  const float* eattr  = (const float*)d_in[2];
  const float* geattr = (const float*)d_in[3];
  const float* gpos   = (const float*)d_in[4];
  auto PW = [&](int i) { return (const float*)d_in[5 + i]; };
  const int* zidx  = (const int*)d_in[55];
  const int* ntype = (const int*)d_in[56];
  const int* eidx  = (const int*)d_in[57];
  const int* geidx = (const int*)d_in[59];

  // --- workspace allocator ---
  char* ws = (char*)d_ws;
  size_t off = 0;
  auto alloc = [&](size_t bytes) -> void* {
    void* p = ws + off;
    off = (off + bytes + 255) & ~(size_t)255;
    return p;
  };

  auto wconv = [&](const float* src, int Co, int Copad, int K0, int Kpad,
                   int colOff, int srcLd) -> bf16* {
    bf16* dst = (bf16*)alloc((size_t)Copad * Kpad * sizeof(bf16));
    int n = Copad * Kpad;
    k_wconv<<<divup(n, 256), 256, 0, stream>>>(dst, src, Co, Copad, K0, Kpad, colOff, srcLd);
    return dst;
  };
  auto zero = [&](void* p, size_t words) {
    k_zero32<<<divup((int)words, 256), 256, 0, stream>>>((unsigned*)p, (int)words);
  };
  auto gemm = [&](const bf16* A1, int K1, const bf16* B1,
                  const bf16* A2, const bf16* B2, int K2,
                  const float* bias, void* out, int M, int N, int validM,
                  int mode, int relu, int nn, int Co) {
    if ((M & 31) == 0 && (N & 31) == 0) {
      int waves = (M / 32) * (N / 32);
      k_gemm2<<<divup(waves * 32, 256), 256, 0, stream>>>(
          A1, K1, B1, A2, B2, K2, bias, out, M, N, validM, mode, relu, nn, Co);
    } else {
      int waves = (M / 16) * (N / 16);
      k_gemm<<<divup(waves * 32, 256), 256, 0, stream>>>(
          A1, K1, B1, A2, B2, K2, bias, out, M, N, validM, mode, relu, nn, Co);
    }
  };

  // --- convert weights to bf16 (K zero-padded to mult of 32) ---
  // global blocks
  bf16* g0m1a = wconv(PW(0), 128, 128, 22, 32, 0, 22);
  bf16* g0m1b = wconv(PW(2), 128, 128, 128, 128, 0, 128);
  bf16* g0m2a = wconv(PW(4), 128, 128, 22, 32, 0, 22);
  bf16* g0m2b = wconv(PW(6), 128, 128, 128, 128, 0, 128);
  bf16* g0skA = wconv(PW(8), 128, 128, 22, 32, 0, 150);
  bf16* g0skB = wconv(PW(8), 128, 128, 128, 128, 22, 150);
  bf16* g1m1a = wconv(PW(10), 128, 128, 128, 128, 0, 128);
  bf16* g1m1b = wconv(PW(12), 128, 128, 128, 128, 0, 128);
  bf16* g1m2a = wconv(PW(14), 128, 128, 128, 128, 0, 128);
  bf16* g1m2b = wconv(PW(16), 128, 128, 128, 128, 0, 128);
  bf16* g1skA = wconv(PW(18), 128, 128, 128, 128, 0, 256);
  bf16* g1skB = wconv(PW(18), 128, 128, 128, 128, 128, 256);
  bf16* gfc1w = wconv(PW(20), 128, 128, 256, 256, 0, 256);
  bf16* gfc2w = wconv(PW(22), 1, 16, 128, 128, 0, 128);   // pad N 1->16
  float* gfc2b = (float*)alloc(16 * sizeof(float));
  k_padf32<<<1, 16, 0, stream>>>(gfc2b, PW(23), 1, 16);
  // local blocks
  bf16* l0m1a = wconv(PW(24), 128, 128, 25, 32, 0, 25);
  bf16* l0m1b = wconv(PW(26), 128, 128, 128, 128, 0, 128);
  bf16* l0m2a = wconv(PW(28), 128, 128, 25, 32, 0, 25);
  bf16* l0m2b = wconv(PW(30), 128, 128, 128, 128, 0, 128);
  bf16* l0skA = wconv(PW(32), 128, 128, 25, 32, 0, 153);
  bf16* l0skB = wconv(PW(32), 128, 128, 128, 128, 25, 153);
  bf16* l1m1a = wconv(PW(34), 128, 128, 128, 128, 0, 128);
  bf16* l1m1b = wconv(PW(36), 128, 128, 128, 128, 0, 128);
  bf16* l1m2a = wconv(PW(38), 128, 128, 128, 128, 0, 128);
  bf16* l1m2b = wconv(PW(40), 128, 128, 128, 128, 0, 128);
  bf16* l1skA = wconv(PW(42), 128, 128, 128, 128, 0, 256);
  bf16* l1skB = wconv(PW(42), 128, 128, 128, 128, 128, 256);
  bf16* lfc1w = wconv(PW(44), 128, 128, 256, 256, 0, 256);
  bf16* lfc2w = wconv(PW(46), 16, 16, 128, 128, 0, 128);
  const float* ntemb = PW(48);
  const float* zemb  = PW(49);

  // --- activation / scratch buffers ---
  float* nf   = (float*)alloc((size_t)NNODES * 19 * sizeof(float));
  float* E5   = (float*)alloc((size_t)PC * 5 * sizeof(float));
  bf16* Z0    = (bf16*)alloc((size_t)MPADL * 32 * sizeof(bf16));
  bf16* ZA    = (bf16*)alloc((size_t)MPADL * 128 * sizeof(bf16));
  bf16* ZB    = (bf16*)alloc((size_t)MPADL * 128 * sizeof(bf16));
  bf16* Hb    = (bf16*)alloc((size_t)MPADL * 128 * sizeof(bf16));
  bf16* M1t   = (bf16*)alloc((size_t)NBC * 128 * 1024 * sizeof(bf16));
  bf16* M2t   = (bf16*)alloc((size_t)NBC * 128 * 1024 * sizeof(bf16));
  bf16* poolL = (bf16*)alloc((size_t)BSUB * 256 * sizeof(bf16));
  bf16* fc1L  = (bf16*)alloc((size_t)BSUB * 128 * sizeof(bf16));
  float* zl   = (float*)alloc((size_t)BSUB * 16 * sizeof(float));
  bf16* Zg0   = (bf16*)alloc((size_t)PG * 32 * sizeof(bf16));
  bf16* ZgA   = (bf16*)alloc((size_t)PG * 128 * sizeof(bf16));
  bf16* ZgB   = (bf16*)alloc((size_t)PG * 128 * sizeof(bf16));
  bf16* Hg    = (bf16*)alloc((size_t)PG * 128 * sizeof(bf16));
  bf16* M1g   = (bf16*)alloc((size_t)GG * 128 * 1024 * sizeof(bf16));
  bf16* M2g   = (bf16*)alloc((size_t)GG * 128 * 1024 * sizeof(bf16));
  bf16* poolG = (bf16*)alloc((size_t)GG * 256 * sizeof(bf16));
  bf16* fc1G  = (bf16*)alloc((size_t)GG * 128 * sizeof(bf16));

  // zero bmm operand pads once (valid region fully rewritten each pass)
  zero(M1t, (size_t)NBC * 128 * 1024 / 2);
  zero(M2t, (size_t)NBC * 128 * 1024 / 2);
  zero(M1g, (size_t)GG * 128 * 1024 / 2);
  zero(M2g, (size_t)GG * 128 * 1024 / 2);

  // node features for local stage
  k_nodefeat<<<divup(NNODES, 256), 256, 0, stream>>>(nf, x, ntemb, zemb, ntype, zidx, NNODES);

  // ---------------- local stage (chunked over subgraphs) ----------------
  for (int ch = 0; ch < NCHUNK; ++ch) {
    int b0 = ch * NBC;
    zero(E5, (size_t)PC * 5);
    k_scatter<<<divup(ELOC, 256), 256, 0, stream>>>(E5, eidx, eattr, ELOC, KK, b0, NBC);
    k_assemble<<<divup(PC, 256), 256, 0, stream>>>(Z0, E5, nf, 19, pos, KK, b0, NBC);

    // block 1 (Cin=25 padded to 32)
    gemm(Z0, 32, l0m1a, 0, 0, 0, PW(25), Hb, MPADL, 128, MPADL, 0, 1, 0, 0);
    gemm(Hb, 128, l0m1b, 0, 0, 0, PW(27), M1t, MPADL, 128, PC, 2, 1, KK, 128);
    gemm(Z0, 32, l0m2a, 0, 0, 0, PW(29), Hb, MPADL, 128, MPADL, 0, 1, 0, 0);
    gemm(Hb, 128, l0m2b, 0, 0, 0, PW(31), M2t, MPADL, 128, PC, 3, 1, KK, 128);
    k_bmm<<<divup(NBC * 128 * 4 * 32, 256), 256, 0, stream>>>(M1t, M2t, Hb, NBC, KK);
    gemm(Z0, 32, l0skA, Hb, l0skB, 128, PW(33), ZA, MPADL, 128, MPADL, 0, 0, 0, 0);

    // block 2 (Cin=128)
    gemm(ZA, 128, l1m1a, 0, 0, 0, PW(35), Hb, MPADL, 128, MPADL, 0, 1, 0, 0);
    gemm(Hb, 128, l1m1b, 0, 0, 0, PW(37), M1t, MPADL, 128, PC, 2, 1, KK, 128);
    gemm(ZA, 128, l1m2a, 0, 0, 0, PW(39), Hb, MPADL, 128, MPADL, 0, 1, 0, 0);
    gemm(Hb, 128, l1m2b, 0, 0, 0, PW(41), M2t, MPADL, 128, PC, 3, 1, KK, 128);
    k_bmm<<<divup(NBC * 128 * 4 * 32, 256), 256, 0, stream>>>(M1t, M2t, Hb, NBC, KK);
    gemm(ZA, 128, l1skA, Hb, l1skB, 128, PW(43), ZB, MPADL, 128, MPADL, 0, 0, 0, 0);

    k_pool<<<divup(NBC * 128 * 32, 256), 256, 0, stream>>>(ZB, poolL, NBC, KK, b0);
  }

  // local heads: fc1 (relu), fc2 (linear, f32 out = subgraph embeddings)
  gemm(poolL, 256, lfc1w, 0, 0, 0, PW(45), fc1L, BSUB, 128, BSUB, 0, 1, 0, 0);
  gemm(fc1L, 128, lfc2w, 0, 0, 0, PW(47), zl, BSUB, 16, BSUB, 1, 0, 0, 0);

  // ---------------- global stage ----------------
  zero(E5, (size_t)PG * 5);
  k_scatter<<<divup(EGLB, 256), 256, 0, stream>>>(E5, geidx, geattr, EGLB, SS, 0, GG);
  k_assemble<<<divup(PG, 256), 256, 0, stream>>>(Zg0, E5, zl, 16, gpos, SS, 0, GG);

  // block 1 (Cin=22 padded to 32)
  gemm(Zg0, 32, g0m1a, 0, 0, 0, PW(1), Hg, PG, 128, PG, 0, 1, 0, 0);
  gemm(Hg, 128, g0m1b, 0, 0, 0, PW(3), M1g, PG, 128, PG, 2, 1, SS, 128);
  gemm(Zg0, 32, g0m2a, 0, 0, 0, PW(5), Hg, PG, 128, PG, 0, 1, 0, 0);
  gemm(Hg, 128, g0m2b, 0, 0, 0, PW(7), M2g, PG, 128, PG, 3, 1, SS, 128);
  k_bmm<<<divup(GG * 128 * 4 * 32, 256), 256, 0, stream>>>(M1g, M2g, Hg, GG, SS);
  gemm(Zg0, 32, g0skA, Hg, g0skB, 128, PW(9), ZgA, PG, 128, PG, 0, 0, 0, 0);

  // block 2 (Cin=128)
  gemm(ZgA, 128, g1m1a, 0, 0, 0, PW(11), Hg, PG, 128, PG, 0, 1, 0, 0);
  gemm(Hg, 128, g1m1b, 0, 0, 0, PW(13), M1g, PG, 128, PG, 2, 1, SS, 128);
  gemm(ZgA, 128, g1m2a, 0, 0, 0, PW(15), Hg, PG, 128, PG, 0, 1, 0, 0);
  gemm(Hg, 128, g1m2b, 0, 0, 0, PW(17), M2g, PG, 128, PG, 3, 1, SS, 128);
  k_bmm<<<divup(GG * 128 * 4 * 32, 256), 256, 0, stream>>>(M1g, M2g, Hg, GG, SS);
  gemm(ZgA, 128, g1skA, Hg, g1skB, 128, PW(19), ZgB, PG, 128, PG, 0, 0, 0, 0);

  k_pool<<<divup(GG * 128 * 32, 256), 256, 0, stream>>>(ZgB, poolG, GG, SS, 0);

  // global heads: fc1 (relu), fc2 (linear, N padded 1->16, write column 0 -> d_out[16])
  gemm(poolG, 256, gfc1w, 0, 0, 0, PW(21), fc1G, GG, 128, GG, 0, 1, 0, 0);
  gemm(fc1G, 128, gfc2w, 0, 0, 0, gfc2b, d_out, GG, 16, GG, 4, 0, 0, 0);
}
